// MultiHeadAttention_6614249636606
// MI455X (gfx1250) — compile-verified, matrix operand order checked by reading, not run on hardware
//
#include <hip/hip_runtime.h>
#include <hip/hip_bf16.h>

typedef __bf16 bf16_t;
typedef __attribute__((ext_vector_type(16))) __bf16 v16bf;
typedef __attribute__((ext_vector_type(8)))  __bf16 v8bf;
typedef __attribute__((ext_vector_type(8)))  float  v8f;

constexpr int    Cc  = 256;
constexpr int    Hh  = 8;
constexpr int    Dd  = 32;   // head dim = exactly one bf16 WMMA K-step
constexpr int    Ll  = 2048;
constexpr int    Nn  = 8;
constexpr size_t NCL = (size_t)Nn * Cc * Ll;   // 4,194,304 elements

__device__ __forceinline__ v8f v8f_zero() {
  v8f z = {0.f, 0.f, 0.f, 0.f, 0.f, 0.f, 0.f, 0.f};
  return z;
}

__device__ __forceinline__ v8f wmma_bf16(v16bf a, v16bf b, v8f c) {
  return __builtin_amdgcn_wmma_f32_16x16x32_bf16(false, a, false, b, (short)0, c,
                                                 false, false);
}

// Load one 16-bit fragment half-pair: halves 0-7 from p[0..7], 8-15 from
// p[16..23].  Both are 16-byte aligned contiguous runs -> 2x global_load_b128
// (or ds_load_b128 when p points into LDS).
union FragU { uint4 u; v8bf h; };
__device__ __forceinline__ v16bf load_frag16(const bf16_t* p) {
  FragU lo, hi;
  lo.u = *(const uint4*)p;
  hi.u = *(const uint4*)(p + 16);
  return __builtin_shufflevector(lo.h, hi.h, 0, 1, 2, 3, 4, 5, 6, 7,
                                 8, 9, 10, 11, 12, 13, 14, 15);
}

// ---------------------------------------------------------------------------
// Kernel 0: transpose+convert x[N,C,L] f32 -> xT[N,L,C] bf16 via LDS tile so
// both global sides are coalesced.  Makes all QKV B-fragments contiguous.
// ---------------------------------------------------------------------------
__global__ void __launch_bounds__(256) transpose_x_kernel(
    const float* __restrict__ x, bf16_t* __restrict__ xT) {
  __shared__ bf16_t tile[32][33];           // +1 pad: conflict-free transpose
  const int b  = blockIdx.x;
  const int lt = b & 63;                    // 64 L tiles of 32
  const int ct = (b >> 6) & 7;              // 8 C tiles of 32
  const int nb = b >> 9;
  const int tid = threadIdx.x;

  {
    const int tl = tid & 31;                // consecutive threads -> l : coalesced
    const int c0 = (tid >> 5) * 4;
#pragma unroll
    for (int i = 0; i < 4; ++i) {
      const int c = c0 + i;
      tile[c][tl] = (bf16_t)x[((size_t)nb * Cc + ct * 32 + c) * Ll + lt * 32 + tl];
    }
  }
  __syncthreads();
  {
    const int tc = tid & 31;                // consecutive threads -> c : contiguous
    const int l0 = (tid >> 5) * 4;
#pragma unroll
    for (int i = 0; i < 4; ++i) {
      const int l = l0 + i;
      xT[((size_t)nb * Ll + lt * 32 + l) * Cc + ct * 32 + tc] = tile[tc][l];
    }
  }
}

// ---------------------------------------------------------------------------
// Kernel 1: fused QKV projection.  One wave -> 16(out-ch) x 64(L) tile of one
// of {Q,K,V}; one A fragment feeds 4 back-to-back WMMAs.  B fragments are
// contiguous b128 loads from xT.  Q/K written position-major [N,H,L,D] (one
// uint4 store per sub-tile), V written channel-major [N,H,D,L].
// ---------------------------------------------------------------------------
__global__ void __launch_bounds__(256) qkv_proj_kernel(
    const bf16_t* __restrict__ xT, const float* __restrict__ Wq,
    const float* __restrict__ Wk,  const float* __restrict__ Wv,
    bf16_t* __restrict__ QT, bf16_t* __restrict__ KT, bf16_t* __restrict__ Vb) {
  const int wg   = blockIdx.x * 8 + (threadIdx.x >> 5);
  const int lane = threadIdx.x & 31;
  const int lt   = wg & 31;             // L tile (64 positions)
  const int ot   = (wg >> 5) & 15;      // out-channel tile (16 rows)
  const int nb   = (wg >> 9) & 7;       // batch
  const int mat  = wg >> 12;            // 0=Q, 1=K, 2=V

  const float* W = (mat == 0) ? Wq : (mat == 1) ? Wk : Wv;

  const int m     = lane & 15;
  const int basek = (lane < 16) ? 0 : 8;
  const int hi8   = (lane < 16) ? 0 : 8;
  const int l0    = lt * 64 + m;

  v8f acc[4];
#pragma unroll
  for (int j = 0; j < 4; ++j) acc[j] = v8f_zero();

  for (int cc = 0; cc < Cc; cc += 32) {
    // A fragment: W rows, f32 -> bf16 (vector f32 loads, packed converts)
    v16bf a;
    {
      const float* wr = W + (size_t)(ot * 16 + m) * Cc + cc + basek;
      union { float4 v[2]; float f[8]; } flo, fhi;
      flo.v[0] = *(const float4*)(wr);
      flo.v[1] = *(const float4*)(wr + 4);
      fhi.v[0] = *(const float4*)(wr + 16);
      fhi.v[1] = *(const float4*)(wr + 20);
#pragma unroll
      for (int i = 0; i < 8; ++i) {
        a[i]     = (bf16_t)flo.f[i];
        a[i + 8] = (bf16_t)fhi.f[i];
      }
    }
    // B fragments: contiguous bf16 from xT
    v16bf b[4];
#pragma unroll
    for (int j = 0; j < 4; ++j)
      b[j] = load_frag16(xT + ((size_t)nb * Ll + l0 + j * 16) * Cc + cc + basek);
#pragma unroll
    for (int j = 0; j < 4; ++j) acc[j] = wmma_bf16(a, b[j], acc[j]);  // A reused 4x
  }

  if (mat < 2) {
    // Q/K: position-major [N, H, L, D]; d contiguous across accumulator rows
    bf16_t* OT = (mat == 0) ? QT : KT;
    const int hq = ot >> 1;
    const int d0 = (ot & 1) * 16 + hi8;
#pragma unroll
    for (int j = 0; j < 4; ++j) {
      FragU st;
#pragma unroll
      for (int r = 0; r < 8; ++r) st.h[r] = (bf16_t)acc[j][r];
      *(uint4*)(OT + ((size_t)(nb * Hh + hq) * Ll + l0 + j * 16) * Dd + d0) = st.u;
    }
  } else {
    // V: channel-major [N, H*D, L]
#pragma unroll
    for (int j = 0; j < 4; ++j)
#pragma unroll
      for (int r = 0; r < 8; ++r)
        Vb[((size_t)nb * Cc + ot * 16 + r + hi8) * Ll + l0 + j * 16] =
            (bf16_t)acc[j][r];
  }
}

// ---------------------------------------------------------------------------
// Kernel 2: flash attention.  One wave -> one (n, h, 32-query tile).
// 64-key chunks: 8 S WMMAs (2 Q x 4 K frags), ONE online-softmax pass
// (shfl chains amortized over 64 keys), 32x64 bf16 P tile in LDS, 8 P*V
// WMMAs.  All global fragments are 2x b128 loads; scores never hit HBM.
// ---------------------------------------------------------------------------
__global__ void __launch_bounds__(128) attn_kernel(
    const bf16_t* __restrict__ QT, const bf16_t* __restrict__ KT,
    const bf16_t* __restrict__ Vb, bf16_t* __restrict__ AO) {
  __shared__ bf16_t sP[4][32 * 64];
  const int wslot = threadIdx.x >> 5;
  const int wg    = blockIdx.x * 4 + wslot;
  const int lane  = threadIdx.x & 31;
  const int qt    = wg & 63;            // 32-query tile
  const int h     = (wg >> 6) & 7;
  const int nb    = wg >> 9;

  const int m     = lane & 15;
  const int basek = (lane < 16) ? 0 : 8;
  const int hi8   = (lane < 16) ? 0 : 8;
  const size_t qkBase = (size_t)(nb * Hh + h) * Ll * Dd;     // [N,H,L,D]
  const size_t vBase  = ((size_t)nb * Cc + h * Dd) * Ll;     // [N,H*D,L]
  const float  scale  = 0.0625f;  // 1/sqrt(C) = 1/16 (reference scales by C)

  // Two Q fragments (A-layout): fixed q row per lane, contiguous d
  v16bf qa[2];
#pragma unroll
  for (int t = 0; t < 2; ++t)
    qa[t] = load_frag16(QT + qkBase + (size_t)(qt * 32 + t * 16 + m) * Dd + basek);

  v8f o[2][2];
#pragma unroll
  for (int t = 0; t < 2; ++t) { o[t][0] = v8f_zero(); o[t][1] = v8f_zero(); }
  float mrow[2][8], lrow[2][8];
#pragma unroll
  for (int t = 0; t < 2; ++t)
#pragma unroll
    for (int r = 0; r < 8; ++r) { mrow[t][r] = -1e30f; lrow[t][r] = 0.f; }

  for (int kc = 0; kc < Ll; kc += 64) {
    // Four K fragments (B-layout): fixed key pos per lane, contiguous d
    const bf16_t* kp = KT + qkBase + (size_t)(kc + m) * Dd + basek;
    v16bf kb[4];
#pragma unroll
    for (int j = 0; j < 4; ++j)
      kb[j] = load_frag16(kp + (size_t)(j * 16) * Dd);

    // S = Q K^T : 8 WMMAs
    v8f s[2][4];
#pragma unroll
    for (int t = 0; t < 2; ++t)
#pragma unroll
      for (int j = 0; j < 4; ++j)
        s[t][j] = wmma_bf16(qa[t], kb[j], v8f_zero());

    // One online-softmax pass over all 64 keys; rows live across 16 lanes
    bf16_t* sp = sP[wslot];
#pragma unroll
    for (int t = 0; t < 2; ++t) {
#pragma unroll
      for (int r = 0; r < 8; ++r) {
        float mx = fmaxf(fmaxf(s[t][0][r], s[t][1][r]),
                         fmaxf(s[t][2][r], s[t][3][r])) * scale;
        mx = fmaxf(mx, __shfl_xor(mx, 1, 32));
        mx = fmaxf(mx, __shfl_xor(mx, 2, 32));
        mx = fmaxf(mx, __shfl_xor(mx, 4, 32));
        mx = fmaxf(mx, __shfl_xor(mx, 8, 32));
        const float mnew = fmaxf(mrow[t][r], mx);
        float p[4], rs = 0.f;
#pragma unroll
        for (int j = 0; j < 4; ++j) {
          p[j] = __expf(s[t][j][r] * scale - mnew);
          rs += p[j];
        }
        rs += __shfl_xor(rs, 1, 32);
        rs += __shfl_xor(rs, 2, 32);
        rs += __shfl_xor(rs, 4, 32);
        rs += __shfl_xor(rs, 8, 32);
        const float f = __expf(mrow[t][r] - mnew);
        lrow[t][r] = lrow[t][r] * f + rs;
        mrow[t][r] = mnew;
        o[t][0][r] *= f;
        o[t][1][r] *= f;
#pragma unroll
        for (int j = 0; j < 4; ++j)
          sp[(t * 16 + r + hi8) * 64 + j * 16 + m] = (bf16_t)p[j];
      }
    }
    __syncthreads();
    // Read P back in A-layout per 32-key window: ds_load_b128 pairs
    v16bf pa[2][2];
#pragma unroll
    for (int t = 0; t < 2; ++t)
#pragma unroll
      for (int u = 0; u < 2; ++u)
        pa[t][u] = load_frag16(sp + (t * 16 + m) * 64 + u * 32 + basek);
    __syncthreads();

    // Four V fragments (B-layout): fixed d per lane, contiguous key pos
    const bf16_t* vp = Vb + vBase + (size_t)m * Ll + kc + basek;
    v16bf vb[2][2];  // [key-window][d-half]
#pragma unroll
    for (int u = 0; u < 2; ++u) {
      vb[u][0] = load_frag16(vp + u * 32);
      vb[u][1] = load_frag16(vp + (size_t)16 * Ll + u * 32);
    }

    __builtin_prefetch(kp + (size_t)64 * Dd, 0, 0);  // next K chunk
    __builtin_prefetch(vp + 64, 0, 0);               // next V chunk

    // O += P V : 8 WMMAs
#pragma unroll
    for (int t = 0; t < 2; ++t)
#pragma unroll
      for (int u = 0; u < 2; ++u) {
        o[t][0] = wmma_bf16(pa[t][u], vb[u][0], o[t][0]);
        o[t][1] = wmma_bf16(pa[t][u], vb[u][1], o[t][1]);
      }
  }

  // Normalize rows; store attention output position-major [N, L, C]
#pragma unroll
  for (int t = 0; t < 2; ++t) {
#pragma unroll
    for (int r = 0; r < 8; ++r) {
      const float  inv  = 1.0f / lrow[t][r];
      const size_t qpos = (size_t)qt * 32 + t * 16 + r + hi8;
      AO[((size_t)nb * Ll + qpos) * Cc + h * Dd + m]      = (bf16_t)(o[t][0][r] * inv);
      AO[((size_t)nb * Ll + qpos) * Cc + h * Dd + 16 + m] = (bf16_t)(o[t][1][r] * inv);
    }
  }
}

// ---------------------------------------------------------------------------
// Kernel 3: output projection Wo * attn_out + bo -> f32 [N,C,L].
// B fragments contiguous from AO[N,L,C]; one Wo fragment feeds 4 WMMAs.
// ---------------------------------------------------------------------------
__global__ void __launch_bounds__(256) out_proj_kernel(
    const bf16_t* __restrict__ AO, const float* __restrict__ Wo,
    const float* __restrict__ bo, float* __restrict__ out) {
  const int wg   = blockIdx.x * 8 + (threadIdx.x >> 5);
  const int lane = threadIdx.x & 31;
  const int lt   = wg & 31;
  const int ot   = (wg >> 5) & 15;
  const int nb   = wg >> 9;

  const int m     = lane & 15;
  const int basek = (lane < 16) ? 0 : 8;
  const int hi8   = (lane < 16) ? 0 : 8;
  const int l0    = lt * 64 + m;

  v8f acc[4];
#pragma unroll
  for (int j = 0; j < 4; ++j) acc[j] = v8f_zero();

  for (int cc = 0; cc < Cc; cc += 32) {
    v16bf a;
    {
      const float* wr = Wo + (size_t)(ot * 16 + m) * Cc + cc + basek;
      union { float4 v[2]; float f[8]; } flo, fhi;
      flo.v[0] = *(const float4*)(wr);
      flo.v[1] = *(const float4*)(wr + 4);
      fhi.v[0] = *(const float4*)(wr + 16);
      fhi.v[1] = *(const float4*)(wr + 20);
#pragma unroll
      for (int i = 0; i < 8; ++i) {
        a[i]     = (bf16_t)flo.f[i];
        a[i + 8] = (bf16_t)fhi.f[i];
      }
    }
    v16bf b[4];
#pragma unroll
    for (int j = 0; j < 4; ++j)
      b[j] = load_frag16(AO + ((size_t)nb * Ll + l0 + j * 16) * Cc + cc + basek);
#pragma unroll
    for (int j = 0; j < 4; ++j) acc[j] = wmma_bf16(a, b[j], acc[j]);
  }
#pragma unroll
  for (int j = 0; j < 4; ++j) {
#pragma unroll
    for (int r = 0; r < 8; ++r) {
      const int o = ot * 16 + r + hi8;
      out[((size_t)nb * Cc + o) * Ll + l0 + j * 16] = acc[j][r] + bo[o];
    }
  }
}

// ---------------------------------------------------------------------------
extern "C" void kernel_launch(void* const* d_in, const int* in_sizes, int n_in,
                              void* d_out, int out_size, void* d_ws, size_t ws_size,
                              hipStream_t stream) {
  const float* x  = (const float*)d_in[0];
  const float* Wq = (const float*)d_in[1];
  const float* Wk = (const float*)d_in[2];
  const float* Wv = (const float*)d_in[3];
  const float* Wo = (const float*)d_in[4];
  const float* bo = (const float*)d_in[5];
  float*       out = (float*)d_out;

  // Workspace (bf16): xT[N,L,C], Q[N,H,L,D], K[N,H,L,D], V[N,H*D,L], AO[N,L,C]
  bf16_t* xT = (bf16_t*)d_ws;
  bf16_t* QT = xT + NCL;
  bf16_t* KT = QT + NCL;
  bf16_t* Vb = KT + NCL;
  bf16_t* AO = Vb + NCL;

  // 8 n * 8 c-tiles * 64 l-tiles = 4096 blocks
  transpose_x_kernel<<<4096, 256, 0, stream>>>(x, xT);
  // 3 mats * 8 n * 16 o-tiles * 32 l-tiles = 12288 waves, 8 waves/block
  qkv_proj_kernel<<<1536, 256, 0, stream>>>(xT, Wq, Wk, Wv, QT, KT, Vb);
  // 8 n * 8 h * 64 q-tiles = 4096 waves, 4 waves/block
  attn_kernel<<<1024, 128, 0, stream>>>(QT, KT, Vb, AO);
  // 8 n * 16 o-tiles * 32 l-tiles = 4096 waves, 8 waves/block
  out_proj_kernel<<<512, 256, 0, stream>>>(AO, Wo, bo, out);
}
